// MultiComponentMPNN_86414741995743
// MI455X (gfx1250) — matrix operand order repaired and checked by reading, not compiled, not run on hardware
//
#include <hip/hip_runtime.h>

typedef _Float16 f16;
typedef __attribute__((ext_vector_type(16))) _Float16 v16h;
typedef __attribute__((ext_vector_type(8)))  _Float16 v8h;
typedef __attribute__((ext_vector_type(8)))  float    v8f;

#define N_ATOMS 60000
#define N_BONDS 150000
#define N_MOLS  2000
#define HIDN    300
#define NLD     304      // padded activation stride (19 * 16)
#define NT2     20       // weight n-tiles padded to 20 (= 320 cols, top tile zero)
#define KCH     5        // k-tiles (of 32) per LDS chunk -> 160 K per chunk

union PackU { unsigned u; f16 h[2]; };

// ---------------------------------------------------------------------------
// Weight conversion: f32 row-major [K,300]  ->  f16 swizzled [KTpad][20][512]
// B-fragment layout (16x16x32 f16 WMMA): lane l holds column n=l&15,
// K values k_local = (l>>4)*16 + idx  (idx = 0..15 within lane).
// Zero-fills k >= K and n >= 300 (pad tiles contribute nothing).
// ---------------------------------------------------------------------------
__global__ __launch_bounds__(256) void conv_w_kernel(const float* __restrict__ W,
                                                     f16* __restrict__ out,
                                                     int K, int KTpad)
{
    int o = blockIdx.x * 256 + threadIdx.x;
    int tot = KTpad * NT2 * 512;
    if (o >= tot) return;
    int blk = o >> 9, pos = o & 511;
    int lane = pos >> 4, idx = pos & 15;
    int kt = blk / NT2, ntile = blk - kt * NT2;
    int nlo = lane & 15, half = lane >> 4;
    int k = kt * 32 + half * 16 + idx;
    int n = ntile * 16 + nlo;
    float v = (k < K && n < HIDN) ? W[k * HIDN + n] : 0.f;
    out[o] = (f16)v;
}

// ---------------------------------------------------------------------------
// Generic gathered-A WMMA GEMM:  out = relu(A @ Wswz (+ bias))
// MODE 0 INIT:   A[r,k] = k<133 ? fAtoms[g0[r]*133+k] : k<147 ? fBonds[r*14+k-133] : 0
// MODE 1 UPDATE: A[r,k] = k<300 ? nei[g0[r]*NLD+k] - msg[g1[r]*NLD+k] : 0
// MODE 2 ATOM:   A[r,k] = k<133 ? fAtoms[r*133+k] : k<433 ? nei[r*NLD+k-133] : 0
// MODE 3 PLAIN:  A[r,k] = k<K   ? hSrc[r*lda+k] : 0
// KTpad is a multiple of KCH; every chunk is full (compile-time loop bounds,
// WMMA is straight-line, no EXEC-affecting guards anywhere near it).
// ---------------------------------------------------------------------------
template<int MODE>
__global__ __launch_bounds__(256) void gemm_kernel(
    const float* __restrict__ fAtoms, const float* __restrict__ fBonds,
    const f16*  __restrict__ hNei,    const f16*  __restrict__ hMsg,
    const int*  __restrict__ g0,      const int*  __restrict__ g1,
    const f16*  __restrict__ wswz,    int KTpad,
    const float* __restrict__ bias,
    int M, int K, int lda,
    float* __restrict__ outF, f16* __restrict__ outH, int ldo)
{
    __shared__ f16 ldsA[4 * KCH * 512] __attribute__((aligned(32)));
    const int tid   = threadIdx.x;
    const int lane  = tid & 31;
    const int wave  = tid >> 5;
    const int mtile = wave & 3;            // 4 m-subtiles of 16 rows
    const int ntBase = (wave >> 2) * 10;   // 2 n-groups x 10 tiles (NT2 = 20)
    const int rowBase = blockIdx.x * 64;

    v8f acc[10] = {};

    const int nChunks = KTpad / KCH;
    for (int c = 0; c < nChunks; ++c) {
        const int kt0 = c * KCH;
        __syncthreads();
        __builtin_prefetch(wswz + (((size_t)kt0 * NT2) << 9), 0, 1);

        // -------- build A chunk into LDS in fragment layout --------
        // 64 rows x 80 f16-pairs (5 ktiles * 16 pairs); 20 pairs per thread
        for (int p = tid; p < 64 * KCH * 16; p += 256) {
            const int row = p / (KCH * 16);
            const int rem = p - row * (KCH * 16);
            const int ktl = rem >> 4;
            const int kl  = (rem & 15) * 2;               // even k_local in tile
            const int k   = (kt0 + ktl) * 32 + kl;        // global k (even)
            const int r   = rowBase + row;
            float v0 = 0.f, v1 = 0.f;
            if constexpr (MODE == 0) {
                if (r < M && k < 147) {
                    const int a = g0[r];
                    v0 = (k < 133) ? fAtoms[a * 133 + k]
                                   : fBonds[r * 14 + (k - 133)];
                    const int k1 = k + 1;
                    v1 = (k1 < 133) ? fAtoms[a * 133 + k1]
                       : (k1 < 147) ? fBonds[r * 14 + (k1 - 133)] : 0.f;
                }
            } else if constexpr (MODE == 1) {
                if (r < M && k < HIDN) {
                    const int a = g0[r], rb = g1[r];
                    PackU un, um;
                    un.u = *(const unsigned*)(hNei + (size_t)a  * NLD + k);
                    um.u = *(const unsigned*)(hMsg + (size_t)rb * NLD + k);
                    v0 = (float)un.h[0] - (float)um.h[0];
                    v1 = (float)un.h[1] - (float)um.h[1];
                }
            } else if constexpr (MODE == 2) {
                if (r < M && k < 433) {
                    v0 = (k < 133) ? fAtoms[r * 133 + k]
                                   : (float)hNei[(size_t)r * NLD + (k - 133)];
                    const int k1 = k + 1;
                    v1 = (k1 < 133) ? fAtoms[r * 133 + k1]
                       : (k1 < 433) ? (float)hNei[(size_t)r * NLD + (k1 - 133)] : 0.f;
                }
            } else {
                if (r < M && k < K) {
                    PackU us;
                    us.u = *(const unsigned*)(hNei + (size_t)r * lda + k);
                    v0 = (float)us.h[0]; v1 = (float)us.h[1];
                }
            }
            // A-fragment position (16-bit A 16x32, contiguous k = 16*lhalf + idx, matching B)
            const int mt = row >> 4, m = row & 15;
            const int lhalf = kl >> 4;
            const int v = (kl & 15) >> 1;
            PackU o; o.h[0] = (f16)v0; o.h[1] = (f16)v1;
            *(unsigned*)&ldsA[((mt * KCH + ktl) << 9) + ((m + (lhalf << 4)) << 4) + (v << 1)] = o.u;
        }
        __syncthreads();

        // -------- straight-line WMMA over this chunk --------
        #pragma unroll
        for (int ktl = 0; ktl < KCH; ++ktl) {
            const v16h af = *(const v16h*)&ldsA[((mtile * KCH + ktl) << 9) + (lane << 4)];
            const f16* bbase = wswz + (((size_t)(kt0 + ktl) * NT2) << 9) + (lane << 4);
            #pragma unroll
            for (int nt = 0; nt < 10; ++nt) {
                const v16h bf = *(const v16h*)(bbase + ((size_t)(ntBase + nt) << 9));
                acc[nt] = __builtin_amdgcn_wmma_f32_16x16x32_f16(
                    false, af, false, bf, (short)0, acc[nt], false, false);
            }
        }
    }

    // -------- epilogue: bias + relu, write f16 or f32 --------
    const int nlo = lane & 15, mhalf = lane >> 4;
    #pragma unroll
    for (int nt = 0; nt < 10; ++nt) {
        const int n = (ntBase + nt) * 16 + nlo;
        if (n < HIDN) {
            const float bv = bias ? bias[n] : 0.f;
            #pragma unroll
            for (int i = 0; i < 8; ++i) {
                const int r = rowBase + mtile * 16 + mhalf * 8 + i;
                if (r < M) {
                    float vv = acc[nt][i] + bv;
                    vv = vv > 0.f ? vv : 0.f;
                    if (outH) outH[(size_t)r * ldo + n] = (f16)vv;
                    else      outF[(size_t)r * ldo + n] = vv;
                }
            }
        }
    }
}

// ---------------------------------------------------------------------------
// nei[a,:] = sum_j msg[a2b[a,j], :]   (f16 in/out, f32 accumulate)
// ---------------------------------------------------------------------------
__global__ __launch_bounds__(256) void gather_sum_kernel(
    const f16* __restrict__ msg, const int* __restrict__ a2b,
    f16* __restrict__ nei, int nAtoms)
{
    int t = blockIdx.x * 256 + threadIdx.x;
    int a = t / 38, kq = t - a * 38;      // 38 * 8 halfs = 304
    if (a >= nAtoms) return;
    float s[8] = {};
    #pragma unroll
    for (int j = 0; j < 6; ++j) {
        int b = a2b[a * 6 + j];
        v8h v = *(const v8h*)(msg + (size_t)b * NLD + kq * 8);
        #pragma unroll
        for (int i = 0; i < 8; ++i) s[i] += (float)v[i];
    }
    v8h r;
    #pragma unroll
    for (int i = 0; i < 8; ++i) r[i] = (f16)s[i];
    *(v8h*)(nei + (size_t)a * NLD + kq * 8) = r;
}

// ---------------------------------------------------------------------------
__global__ __launch_bounds__(256) void seg_kernel(
    const float* __restrict__ atomh, const int* __restrict__ mids,
    float* __restrict__ sums, float* __restrict__ cnt, int nAtoms)
{
    int t = blockIdx.x * 256 + threadIdx.x;
    if (t >= nAtoms * HIDN) return;
    int a = t / HIDN, k = t - a * HIDN;
    int m = mids[a];
    __hip_atomic_fetch_add(&sums[(size_t)m * NLD + k], atomh[(size_t)a * NLD + k],
                           __ATOMIC_RELAXED, __HIP_MEMORY_SCOPE_AGENT);
    if (k == 0)
        __hip_atomic_fetch_add(&cnt[m], 1.0f, __ATOMIC_RELAXED, __HIP_MEMORY_SCOPE_AGENT);
}

// ---------------------------------------------------------------------------
__global__ __launch_bounds__(256) void combine_kernel(
    const float* __restrict__ sums_il, const float* __restrict__ cnt_il,
    const float* __restrict__ sums_hl, const float* __restrict__ cnt_hl,
    const float* __restrict__ tab, f16* __restrict__ comb)
{
    int t = blockIdx.x * 256 + threadIdx.x;
    if (t >= N_MOLS * 640) return;
    int m = t / 640, c = t - m * 640;
    float v = 0.f;
    if (c < 300)      v = sums_il[(size_t)m * NLD + c]         / fmaxf(cnt_il[m], 1.0f);
    else if (c < 600) v = sums_hl[(size_t)m * NLD + (c - 300)] / fmaxf(cnt_hl[m], 1.0f);
    else if (c < 616) v = tab[m * 16 + (c - 600)];
    comb[t] = (f16)v;
}

// ---------------------------------------------------------------------------
__global__ __launch_bounds__(256) void head_kernel(
    const f16* __restrict__ fp, const float* __restrict__ w,
    const float* __restrict__ b, float* __restrict__ out)
{
    int m = blockIdx.x * 256 + threadIdx.x;
    if (m >= N_MOLS) return;
    float s = 0.f;
    for (int k = 0; k < HIDN; ++k) s += (float)fp[(size_t)m * NLD + k] * w[k];
    out[m] = s + b[0];
}

// ---------------------------------------------------------------------------
__global__ __launch_bounds__(256) void zero_kernel(float* p, int n)
{
    int t = blockIdx.x * 256 + threadIdx.x;
    if (t < n) p[t] = 0.f;
}

// ---------------------------------------------------------------------------
extern "C" void kernel_launch(void* const* d_in, const int* in_sizes, int n_in,
                              void* d_out, int out_size, void* d_ws, size_t ws_size,
                              hipStream_t stream)
{
    (void)in_sizes; (void)n_in; (void)out_size; (void)ws_size;

    const float* f_atoms[2] = { (const float*)d_in[0],  (const float*)d_in[10] };
    const float* f_bonds[2] = { (const float*)d_in[1],  (const float*)d_in[11] };
    const int*   a2b[2]     = { (const int*)  d_in[2],  (const int*)  d_in[12] };
    const int*   b2a[2]     = { (const int*)  d_in[3],  (const int*)  d_in[13] };
    const int*   b2revb[2]  = { (const int*)  d_in[4],  (const int*)  d_in[14] };
    const int*   mids[2]    = { (const int*)  d_in[5],  (const int*)  d_in[15] };
    const float* Wi[2]      = { (const float*)d_in[6],  (const float*)d_in[16] };
    const float* Wh[2]      = { (const float*)d_in[7],  (const float*)d_in[17] };
    const float* Wo[2]      = { (const float*)d_in[8],  (const float*)d_in[18] };
    const float* Wob[2]     = { (const float*)d_in[9],  (const float*)d_in[19] };
    const float* tab  = (const float*)d_in[20];
    const float* fw1  = (const float*)d_in[21];
    const float* fb1  = (const float*)d_in[22];
    const float* fw2  = (const float*)d_in[23];
    const float* fb2  = (const float*)d_in[24];
    const float* hw   = (const float*)d_in[25];
    const float* hb   = (const float*)d_in[26];

    // k-tile counts padded to multiples of KCH (=5)
    const int KTi = 5, KTh = 10, KTo = 15, KTf1 = 20, KTf2 = 10;
    const size_t SZ_WI = (size_t)KTi  * NT2 * 512 * 2;
    const size_t SZ_WH = (size_t)KTh  * NT2 * 512 * 2;
    const size_t SZ_WO = (size_t)KTo  * NT2 * 512 * 2;
    const size_t SZ_F1 = (size_t)KTf1 * NT2 * 512 * 2;
    const size_t SZ_F2 = (size_t)KTf2 * NT2 * 512 * 2;

    char* ws = (char*)d_ws;
    size_t off = 0;
    auto take = [&](size_t bytes) { size_t r = off; off += (bytes + 255) & ~(size_t)255; return r; };
    size_t o_wi[2], o_wh[2], o_wo[2];
    o_wi[0] = take(SZ_WI); o_wh[0] = take(SZ_WH); o_wo[0] = take(SZ_WO);
    o_wi[1] = take(SZ_WI); o_wh[1] = take(SZ_WH); o_wo[1] = take(SZ_WO);
    size_t o_f1   = take(SZ_F1);
    size_t o_f2   = take(SZ_F2);
    size_t o_sums = take((size_t)N_MOLS * NLD * 4 * 2 + 8192 * 2); // sums_il|sums_hl|cnt_il|cnt_hl
    size_t o_comb = take((size_t)N_MOLS * 640 * 2);
    size_t o_h1   = take((size_t)N_MOLS * NLD * 2);
    size_t o_fp   = take((size_t)N_MOLS * NLD * 2);
    size_t o_msgA = take((size_t)N_BONDS * NLD * 2);
    size_t o_msgB = take((size_t)N_BONDS * NLD * 2);
    size_t o_nei  = take((size_t)N_ATOMS * NLD * 2);

    float* sums[2] = { (float*)(ws + o_sums),
                       (float*)(ws + o_sums + (size_t)N_MOLS * NLD * 4) };
    float* cnt[2]  = { (float*)(ws + o_sums + (size_t)N_MOLS * NLD * 4 * 2),
                       (float*)(ws + o_sums + (size_t)N_MOLS * NLD * 4 * 2 + 8192) };
    f16* comb  = (f16*)(ws + o_comb);
    f16* h1    = (f16*)(ws + o_h1);
    f16* fpbuf = (f16*)(ws + o_fp);
    f16* msgA  = (f16*)(ws + o_msgA);
    f16* msgB  = (f16*)(ws + o_msgB);
    f16* nei   = (f16*)(ws + o_nei);
    float* atomh = (float*)(ws + o_msgB);   // alias: msgB is dead by readout time

    // zero the segment accumulators (ws is poisoned)
    {
        int nz = (int)(((size_t)N_MOLS * NLD * 4 * 2 + 8192 * 2) / 4);
        zero_kernel<<<(nz + 255) / 256, 256, 0, stream>>>((float*)(ws + o_sums), nz);
    }

    // weight conversion to swizzled f16
    for (int cmp = 0; cmp < 2; ++cmp) {
        conv_w_kernel<<<KTi * 40, 256, 0, stream>>>(Wi[cmp], (f16*)(ws + o_wi[cmp]), 147, KTi);
        conv_w_kernel<<<KTh * 40, 256, 0, stream>>>(Wh[cmp], (f16*)(ws + o_wh[cmp]), 300, KTh);
        conv_w_kernel<<<KTo * 40, 256, 0, stream>>>(Wo[cmp], (f16*)(ws + o_wo[cmp]), 433, KTo);
    }
    conv_w_kernel<<<KTf1 * 40, 256, 0, stream>>>(fw1, (f16*)(ws + o_f1), 616, KTf1);
    conv_w_kernel<<<KTf2 * 40, 256, 0, stream>>>(fw2, (f16*)(ws + o_f2), 300, KTf2);

    const int gbBond = (N_BONDS + 63) / 64;
    const int gbAtom = (N_ATOMS + 63) / 64;
    const int gbGath = (N_ATOMS * 38 + 255) / 256;

    for (int cmp = 0; cmp < 2; ++cmp) {
        // msg0 = relu([f_atoms[b2a], f_bonds] @ W_i)
        gemm_kernel<0><<<gbBond, 256, 0, stream>>>(
            f_atoms[cmp], f_bonds[cmp], nullptr, nullptr, b2a[cmp], nullptr,
            (const f16*)(ws + o_wi[cmp]), KTi, nullptr,
            N_BONDS, 147, 0, nullptr, msgA, NLD);
        // two update steps (DEPTH-1), ping-pong msgA <-> msgB
        for (int d = 0; d < 2; ++d) {
            f16* src = d ? msgB : msgA;
            f16* dst = d ? msgA : msgB;
            gather_sum_kernel<<<gbGath, 256, 0, stream>>>(src, a2b[cmp], nei, N_ATOMS);
            gemm_kernel<1><<<gbBond, 256, 0, stream>>>(
                nullptr, nullptr, nei, src, b2a[cmp], b2revb[cmp],
                (const f16*)(ws + o_wh[cmp]), KTh, nullptr,
                N_BONDS, 300, 0, nullptr, dst, NLD);
        }
        // a_msg, readout, segment mean accumulation
        gather_sum_kernel<<<gbGath, 256, 0, stream>>>(msgA, a2b[cmp], nei, N_ATOMS);
        gemm_kernel<2><<<gbAtom, 256, 0, stream>>>(
            f_atoms[cmp], nullptr, nei, nullptr, nullptr, nullptr,
            (const f16*)(ws + o_wo[cmp]), KTo, Wob[cmp],
            N_ATOMS, 433, 0, atomh, nullptr, NLD);
        seg_kernel<<<(N_ATOMS * HIDN + 255) / 256, 256, 0, stream>>>(
            atomh, mids[cmp], sums[cmp], cnt[cmp], N_ATOMS);
    }

    // combined = [mv_il | mv_hl | tabular] as f16 (pad to 640)
    combine_kernel<<<(N_MOLS * 640 + 255) / 256, 256, 0, stream>>>(
        sums[0], cnt[0], sums[1], cnt[1], tab, comb);

    // FFN
    const int gbMol = (N_MOLS + 63) / 64;
    gemm_kernel<3><<<gbMol, 256, 0, stream>>>(
        nullptr, nullptr, comb, nullptr, nullptr, nullptr,
        (const f16*)(ws + o_f1), KTf1, fb1,
        N_MOLS, 616, 640, nullptr, h1, NLD);
    gemm_kernel<3><<<gbMol, 256, 0, stream>>>(
        nullptr, nullptr, h1, nullptr, nullptr, nullptr,
        (const f16*)(ws + o_f2), KTf2, fb2,
        N_MOLS, 300, NLD, nullptr, fpbuf, NLD);

    head_kernel<<<(N_MOLS + 255) / 256, 256, 0, stream>>>(fpbuf, hw, hb, (float*)d_out);
}
